// RNNWithMixedProjectionSimpleTransition_54425825575657
// MI455X (gfx1250) — matrix operand order repaired and checked by reading, not compiled, not run on hardware
//
#include <hip/hip_runtime.h>
#include <math.h>

// ---------------------------------------------------------------------------
// MI455X (gfx1250) implementation.
// Dominant cost: vocab GEMM 2048x32000x1024 (~134 GFLOP) -> compute bound
// => bf16 WMMA (v_wmma_f32_16x16x32_bf16) for all big GEMMs, with
// gfx1250 async global->LDS copies (ASYNCcnt) double-buffered so tile k+1
// streams in while tile k runs on the matrix pipe.
// ---------------------------------------------------------------------------

#define B_   64
#define E_   1024
#define M_   16
#define NS_  32
#define G_   32000
#define CH_  (G_ / 256)   // 125 column chunks in the vocab GEMM

typedef unsigned int  u32;
typedef unsigned short u16;
typedef __attribute__((ext_vector_type(16))) __bf16 v16bf;
typedef __attribute__((ext_vector_type(8)))  float  v8f;
typedef __attribute__((ext_vector_type(4)))  u32    u32x4;
typedef __attribute__((ext_vector_type(2)))  u32    u32x2;

union FragBF { v16bf v; u32x4 q[2]; };

static __device__ inline u16 f2bf(float f) {
    union { float f; u32 u; } x; x.f = f;
    u32 r = x.u + 0x7FFFu + ((x.u >> 16) & 1u);
    return (u16)(r >> 16);
}
static __device__ inline float bf2f(u16 h) {
    union { u32 u; float f; } x; x.u = ((u32)h) << 16;
    return x.f;
}

// Flat shared pointer -> raw LDS byte offset (ISA: LDS aperture uses addr[31:0]).
static __device__ inline u32 lds_off(const void* p) {
    return (u32)(uintptr_t)p;
}
// gfx1250 async global->LDS copies; tracked with ASYNCcnt.
static __device__ inline void async_ld_b64(u32 lds, const void* g) {
    asm volatile("global_load_async_to_lds_b64 %0, %1, off"
                 :: "v"(lds), "v"(g) : "memory");
}
static __device__ inline void async_ld_b128(u32 lds, const void* g) {
    asm volatile("global_load_async_to_lds_b128 %0, %1, off"
                 :: "v"(lds), "v"(g) : "memory");
}
template <int N>
static __device__ inline void wait_async() {
    asm volatile("s_wait_asynccnt %0" :: "i"(N) : "memory");
}

static __device__ inline float blockReduceSum(float* red, float v) {
    red[threadIdx.x] = v; __syncthreads();
    for (int s = blockDim.x >> 1; s > 0; s >>= 1) {
        if ((int)threadIdx.x < s) red[threadIdx.x] += red[threadIdx.x + s];
        __syncthreads();
    }
    float r = red[0]; __syncthreads();
    return r;
}

// ---------------------------------------------------------------------------
// 1) Gather + stdnorm + noise:  lat = stdnorm(latent[zi]) + (noise-0.5)*0.05
// ---------------------------------------------------------------------------
__global__ __launch_bounds__(256) void prep_kernel(
    const int* __restrict__ zi, const float* __restrict__ noise,
    const float* __restrict__ latent,
    float* __restrict__ z0f, float* __restrict__ zf, u16* __restrict__ z0b)
{
    __shared__ float red[256];
    int b = blockIdx.x;
    const float* lat = latent + (size_t)zi[b] * (2 * E_);
    for (int h = 0; h < 2; ++h) {
        const float* x = lat + h * E_;
        float s = 0.f, s2 = 0.f;
        for (int e = threadIdx.x; e < E_; e += 256) { float v = x[e]; s += v; s2 += v * v; }
        float S  = blockReduceSum(red, s);
        float S2 = blockReduceSum(red, s2);
        float mean = S * (1.f / E_);
        float var  = fmaxf((S2 - (float)E_ * mean * mean) * (1.f / (E_ - 1)), 0.f);
        float scale = 0.113f / (1e-5f + sqrtf(var));
        for (int e = threadIdx.x; e < E_; e += 256) {
            float v = x[e] * scale + (noise[((size_t)b * 2 + h) * E_ + e] - 0.5f) * 0.05f;
            if (h == 0) { z0f[(size_t)b * E_ + e] = v; z0b[(size_t)b * E_ + e] = f2bf(v); }
            else        { zf [(size_t)b * E_ + e] = v; }
        }
        __syncthreads();
    }
}

// ---------------------------------------------------------------------------
// 2) f32 -> bf16 conversion (streaming)
// ---------------------------------------------------------------------------
__global__ void cvt_kernel(const float* __restrict__ x, u16* __restrict__ o, size_t n)
{
    size_t i = (size_t)blockIdx.x * blockDim.x + threadIdx.x;
    size_t stride = (size_t)gridDim.x * blockDim.x;
    for (; i < n; i += stride) o[i] = f2bf(x[i]);
}

// ---------------------------------------------------------------------------
// WMMA GEMM core, templated on RT row-tiles (tile = RT*64 rows x 256 cols).
// 512 threads = 16 waves laid out 4(row) x 4(col); each wave owns RT x 16x64.
// Async double-buffered staging: per K-step of 32 each wave issues 3 async
// global->LDS ops for tile k+1, waits ASYNCcnt<=3 (tile k complete), then
// feeds the matrix pipe from LDS.
// Fragment layouts per CDNA5 ISA 7.12.2:
//   A: lane -> M = lane%16, half lane>>4 selects K {0..7,16..23}/{8..15,24..31}
//   B: lane -> N = lane%16, half selects K {0..7,16..23}/{8..15,24..31} (same as A)
//   C/D: lane -> N = lane%16, VGPR v -> M = v + 8*(lane>>4)
// ---------------------------------------------------------------------------
template <int RT>
static __device__ inline void gemm_core(
    const u16* __restrict__ A, const u16* __restrict__ W, int K,
    int rowBase, int colBase, u16* sA, u16* sB, v8f (&acc)[RT][4])
{
    const int tid = threadIdx.x, lane = tid & 31, wave = tid >> 5;
    const int wr = wave >> 2, wc = wave & 3;
    const int nk = K >> 5;
    const int ABUF = RT * 64 * 32;   // A elems per buffer
    const int BBUF = 256 * 32;       // B elems per buffer

    // per-thread staging coordinates (uniform across waves, all lanes active)
    int a_row, a_k;
    if constexpr (RT == 2) { int idx = tid * 8; a_row = idx >> 5; a_k = idx & 31; }
    else                   { int idx = tid * 4; a_row = idx >> 5; a_k = idx & 31; }
    const int b_col = tid >> 1, b_k = (tid & 1) * 16;

    auto issue_tile = [&](int kb, int buf) {
        const u16* ga = A + (size_t)(rowBase + a_row) * K + kb * 32 + a_k;
        if constexpr (RT == 2)
            async_ld_b128(lds_off(&sA[buf * ABUF + a_row * 32 + a_k]), ga);
        else
            async_ld_b64(lds_off(&sA[buf * ABUF + a_row * 32 + a_k]), ga);
        const u16* gb = W + (size_t)(colBase + b_col) * K + kb * 32 + b_k;
        async_ld_b128(lds_off(&sB[buf * BBUF + b_col * 32 + b_k]),     gb);
        async_ld_b128(lds_off(&sB[buf * BBUF + b_col * 32 + b_k + 8]), gb + 8);
    };

    issue_tile(0, 0);
    for (int kb = 0; kb < nk; ++kb) {
        const int cur = kb & 1;
        if (kb + 1 < nk) { issue_tile(kb + 1, cur ^ 1); wait_async<3>(); }
        else             { wait_async<0>(); }
        __syncthreads();   // all waves' tile-kb data resident in LDS

        FragBF fa[RT];
        #pragma unroll
        for (int rt = 0; rt < RT; ++rt) {
            int ar = rt * 64 + wr * 16 + (lane & 15);
            int ab = 8 * (lane >> 4);
            fa[rt].q[0] = *(const u32x4*)&sA[cur * ABUF + ar * 32 + ab];
            fa[rt].q[1] = *(const u32x4*)&sA[cur * ABUF + ar * 32 + ab + 16];
        }
        FragBF fb[4];
        #pragma unroll
        for (int j = 0; j < 4; ++j) {
            int bc = wc * 64 + j * 16 + (lane & 15);
            int bk = 8 * (lane >> 4);
            fb[j].q[0] = *(const u32x4*)&sB[cur * BBUF + bc * 32 + bk];
            fb[j].q[1] = *(const u32x4*)&sB[cur * BBUF + bc * 32 + bk + 16];
        }
        #pragma unroll
        for (int j = 0; j < 4; ++j)
            #pragma unroll
            for (int rt = 0; rt < RT; ++rt)
                acc[rt][j] = __builtin_amdgcn_wmma_f32_16x16x32_bf16(
                    false, fa[rt].v, false, fb[j].v, (short)0, acc[rt][j], false, false);
        __syncthreads();   // all reads done before buffer cur is refilled
    }
}

// ---------------------------------------------------------------------------
// 3) Plain GEMM + bias (ek / ev projections: 64 x 16384, K=1024)
// ---------------------------------------------------------------------------
__global__ __launch_bounds__(512) void gemm_bias_kernel(
    const u16* __restrict__ A, const u16* __restrict__ W,
    const float* __restrict__ bias, float* __restrict__ C, int K, int cols)
{
    __shared__ u16 sA[2 * 64 * 32];
    __shared__ u16 sB[2 * 256 * 32];
    int rowBase = blockIdx.y * 64, colBase = blockIdx.x * 256;
    v8f acc[1][4];
    #pragma unroll
    for (int j = 0; j < 4; ++j)
        #pragma unroll
        for (int i = 0; i < 8; ++i) acc[0][j][i] = 0.f;
    gemm_core<1>(A, W, K, rowBase, colBase, sA, sB, acc);

    const int lane = threadIdx.x & 31, wave = threadIdx.x >> 5;
    const int wr = wave >> 2, wc = wave & 3;
    const int half = lane >> 4, n = lane & 15;
    #pragma unroll
    for (int j = 0; j < 4; ++j) {
        int col = colBase + wc * 64 + j * 16 + n;
        float bb = bias[col];
        #pragma unroll
        for (int v = 0; v < 8; ++v) {
            int row = rowBase + wr * 16 + v + 8 * half;
            C[(size_t)row * cols + col] = acc[0][j][v] + bb;
        }
    }
}

// ---------------------------------------------------------------------------
// 4) Row stdnorm over 1024-wide rows; writes f32 and/or bf16 output
// ---------------------------------------------------------------------------
__global__ __launch_bounds__(256) void stdnorm_kernel(
    const float* __restrict__ raw, float* __restrict__ outf, u16* __restrict__ outb)
{
    __shared__ float red[256];
    size_t r = blockIdx.x;
    const float* x = raw + r * E_;
    float s = 0.f, s2 = 0.f;
    for (int e = threadIdx.x; e < E_; e += 256) { float v = x[e]; s += v; s2 += v * v; }
    float S  = blockReduceSum(red, s);
    float S2 = blockReduceSum(red, s2);
    float mean = S * (1.f / E_);
    float var  = fmaxf((S2 - (float)E_ * mean * mean) * (1.f / (E_ - 1)), 0.f);
    float scale = 0.113f / (1e-5f + sqrtf(var));
    for (int e = threadIdx.x; e < E_; e += 256) {
        float v = x[e] * scale;
        if (outf) outf[r * E_ + e] = v;
        if (outb) outb[r * E_ + e] = f2bf(v);
    }
}

// ---------------------------------------------------------------------------
// 5) Recurrence: 4 workgroups x 16 batch rows, 32 steps internal.
//    Per step: wave-level stdnorm, attention over M=16, softmax, emit -> zs,
//    z_next = (zn @ trans_w.T + b)/2 via WMMA (wave w owns cols [64w,64w+64),
//    B fragments streamed from L2, batched 8 loads -> 4 WMMAs per K-step).
// ---------------------------------------------------------------------------
__global__ __launch_bounds__(512) void recurrence_kernel(
    float* __restrict__ zbuf,            // (64,1024) f32 state, in/out
    const u16* __restrict__ ekb,         // (64*16,1024) bf16 normalized keys
    const float* __restrict__ evf,       // (64*16,1024) f32 normalized values
    const u16* __restrict__ transb,      // (1024,1024) bf16 trans_w [out][in]
    const float* __restrict__ trans_bias,
    float* __restrict__ zs_f, u16* __restrict__ zs_b)  // (2048,1024)
{
    __shared__ u16   s_znb[16 * E_];     // bf16 zn for this 16-row block (32KB)
    __shared__ float s_att[16][16];
    const int tid = threadIdx.x, lane = tid & 31, wave = tid >> 5;
    const int bBase = blockIdx.x * 16;

    for (int t = 0; t < NS_; ++t) {
        // --- phase 1: zn = stdnorm(z); one wave per row ---
        {
            const float* zr = zbuf + (size_t)(bBase + wave) * E_;
            float s = 0.f, s2 = 0.f;
            for (int e = lane; e < E_; e += 32) { float v = zr[e]; s += v; s2 += v * v; }
            #pragma unroll
            for (int m = 1; m < 32; m <<= 1) {
                s  += __shfl_xor(s,  m, 32);
                s2 += __shfl_xor(s2, m, 32);
            }
            float mean = s * (1.f / E_);
            float var  = fmaxf((s2 - (float)E_ * mean * mean) * (1.f / (E_ - 1)), 0.f);
            float scale = 0.113f / (1e-5f + sqrtf(var));
            for (int e = lane; e < E_; e += 32)
                s_znb[wave * E_ + e] = f2bf(zr[e] * scale);
        }
        __syncthreads();
        // --- phase 2: attention logits; wave w handles b = w, all 16 m ---
        {
            int b = wave;
            for (int m = 0; m < 16; ++m) {
                const u16* ek = ekb + (size_t)((bBase + b) * 16 + m) * E_;
                float a = 0.f;
                for (int e = lane; e < E_; e += 32)
                    a += bf2f(s_znb[b * E_ + e]) * bf2f(ek[e]);
                #pragma unroll
                for (int mm = 1; mm < 32; mm <<= 1) a += __shfl_xor(a, mm, 32);
                if (lane == 0) s_att[b][m] = a;
            }
        }
        __syncthreads();
        // --- phase 3: softmax over m ---
        if (wave == 0 && lane < 16) {
            float mx = -3.4e38f;
            for (int m = 0; m < 16; ++m) mx = fmaxf(mx, s_att[lane][m]);
            float ex[16]; float sm = 0.f;
            for (int m = 0; m < 16; ++m) { ex[m] = expf(s_att[lane][m] - mx); sm += ex[m]; }
            float inv = 1.f / sm;
            for (int m = 0; m < 16; ++m) s_att[lane][m] = ex[m] * inv;
        }
        __syncthreads();
        // --- phase 4: emit -> zs[:, t] ---
        {
            int b = wave;
            for (int e = lane; e < E_; e += 32) {
                float acc = 0.f;
                for (int m = 0; m < 16; ++m)
                    acc += s_att[b][m] * evf[(size_t)((bBase + b) * 16 + m) * E_ + e];
                size_t row = (size_t)(bBase + b) * NS_ + t;
                zs_f[row * E_ + e] = acc;
                zs_b[row * E_ + e] = f2bf(acc);
            }
        }
        // --- phase 5: z_next = (zn @ trans_w.T)/2 + trans_b/2, WMMA ---
        {
            v8f acc[4];
            #pragma unroll
            for (int j = 0; j < 4; ++j)
                #pragma unroll
                for (int i = 0; i < 8; ++i) acc[j][i] = 0.f;
            for (int kb = 0; kb < (E_ >> 5); ++kb) {
                FragBF fa;
                int ar = lane & 15;
                int ab = 8 * (lane >> 4);
                fa.q[0] = *(const u32x4*)&s_znb[ar * E_ + kb * 32 + ab];
                fa.q[1] = *(const u32x4*)&s_znb[ar * E_ + kb * 32 + ab + 16];
                FragBF fb[4];
                #pragma unroll
                for (int j = 0; j < 4; ++j) {
                    int col = wave * 64 + j * 16 + (lane & 15);
                    int k   = kb * 32 + 8 * (lane >> 4);
                    const u16* wp = transb + (size_t)col * E_ + k;
                    fb[j].q[0] = *(const u32x4*)wp;
                    fb[j].q[1] = *(const u32x4*)(wp + 16);
                }
                #pragma unroll
                for (int j = 0; j < 4; ++j)
                    acc[j] = __builtin_amdgcn_wmma_f32_16x16x32_bf16(
                        false, fa.v, false, fb[j].v, (short)0, acc[j], false, false);
            }
            const int half = lane >> 4, n = lane & 15;
            #pragma unroll
            for (int j = 0; j < 4; ++j) {
                int col = wave * 64 + j * 16 + n;
                float bb = 0.5f * trans_bias[col];
                #pragma unroll
                for (int v = 0; v < 8; ++v) {
                    int b = v + 8 * half;
                    zbuf[(size_t)(bBase + b) * E_ + col] = 0.5f * acc[j][v] + bb;
                }
            }
        }
        __threadfence();
        __syncthreads();
    }
}

// ---------------------------------------------------------------------------
// 6) Vocab GEMM (2048 x 32000, K=1024) with fused log-softmax partials.
//    128-row x 256-col workgroup tiles (RT=2): halves L2 weight traffic and
//    runs 8 back-to-back WMMAs per K-step per wave.
// ---------------------------------------------------------------------------
__global__ __launch_bounds__(512) void vocab_kernel(
    const u16* __restrict__ A,            // zs bf16 (2048,1024)
    const u16* __restrict__ W,            // vocab bf16 (32000,1024)
    const float* __restrict__ vb,         // vocab bias
    float* __restrict__ partM, float* __restrict__ partS)
{
    __shared__ u16 sA[2 * 128 * 32];
    __shared__ u16 sB[2 * 256 * 32];
    __shared__ float s_red[4][16][4];
    int rowBase = blockIdx.y * 128, colBase = blockIdx.x * 256;
    int chunk = blockIdx.x;

    v8f acc[2][4];
    #pragma unroll
    for (int rt = 0; rt < 2; ++rt)
        #pragma unroll
        for (int j = 0; j < 4; ++j)
            #pragma unroll
            for (int i = 0; i < 8; ++i) acc[rt][j][i] = 0.f;
    gemm_core<2>(A, W, E_, rowBase, colBase, sA, sB, acc);

    const int lane = threadIdx.x & 31, wave = threadIdx.x >> 5;
    const int wr = wave >> 2, wc = wave & 3;
    const int half = lane >> 4, n = lane & 15;

    for (int rt = 0; rt < 2; ++rt) {
        // logits = acc + bias
        float l[4][8];
        #pragma unroll
        for (int j = 0; j < 4; ++j) {
            float bb = vb[colBase + wc * 64 + j * 16 + n];
            #pragma unroll
            for (int v = 0; v < 8; ++v) l[j][v] = acc[rt][j][v] + bb;
        }
        // per-row max over this wave's 64 columns (reduce across 16 lanes)
        float rmax[8];
        #pragma unroll
        for (int v = 0; v < 8; ++v) {
            float m = fmaxf(fmaxf(l[0][v], l[1][v]), fmaxf(l[2][v], l[3][v]));
            #pragma unroll
            for (int mask = 1; mask < 16; mask <<= 1) m = fmaxf(m, __shfl_xor(m, mask, 32));
            rmax[v] = m;
        }
        __syncthreads();
        if (n == 0)
            #pragma unroll
            for (int v = 0; v < 8; ++v) s_red[wr][v + 8 * half][wc] = rmax[v];
        __syncthreads();
        float rM[8];
        #pragma unroll
        for (int v = 0; v < 8; ++v) {
            int rr = v + 8 * half;
            rM[v] = fmaxf(fmaxf(s_red[wr][rr][0], s_red[wr][rr][1]),
                          fmaxf(s_red[wr][rr][2], s_red[wr][rr][3]));
        }
        // per-row sum of exp over this wave's 64 columns
        float rsum[8];
        #pragma unroll
        for (int v = 0; v < 8; ++v) {
            float s = 0.f;
            #pragma unroll
            for (int j = 0; j < 4; ++j) s += expf(l[j][v] - rM[v]);
            #pragma unroll
            for (int mask = 1; mask < 16; mask <<= 1) s += __shfl_xor(s, mask, 32);
            rsum[v] = s;
        }
        __syncthreads();
        if (n == 0)
            #pragma unroll
            for (int v = 0; v < 8; ++v) s_red[wr][v + 8 * half][wc] = rsum[v];
        __syncthreads();
        if (wc == 0 && n == 0) {
            #pragma unroll
            for (int v = 0; v < 8; ++v) {
                int rr = v + 8 * half;
                int row = rowBase + rt * 64 + wr * 16 + rr;
                float S = s_red[wr][rr][0] + s_red[wr][rr][1] +
                          s_red[wr][rr][2] + s_red[wr][rr][3];
                partM[(size_t)row * CH_ + chunk] = rM[v];
                partS[(size_t)row * CH_ + chunk] = S;
            }
        }
        __syncthreads();
    }
}

// ---------------------------------------------------------------------------
// 7) Finalize: combine chunk partials -> logsumexp; target logit in f32;
//    yp[b,t] = logit(y) - lse
// ---------------------------------------------------------------------------
__global__ __launch_bounds__(256) void finalize_kernel(
    const float* __restrict__ zs, const float* __restrict__ vw,
    const float* __restrict__ vb, const int* __restrict__ y,
    const float* __restrict__ partM, const float* __restrict__ partS,
    float* __restrict__ yp)
{
    __shared__ float red[256];
    int r = blockIdx.x;
    int g = y[r];                          // y is (B,NS) row-major == r
    // target logit in f32
    float d = 0.f;
    for (int e = threadIdx.x; e < E_; e += 256)
        d += zs[(size_t)r * E_ + e] * vw[(size_t)g * E_ + e];
    red[threadIdx.x] = d; __syncthreads();
    for (int s = 128; s > 0; s >>= 1) {
        if ((int)threadIdx.x < s) red[threadIdx.x] += red[threadIdx.x + s];
        __syncthreads();
    }
    float logit = red[0] + vb[g];
    __syncthreads();
    // global max over chunks
    float m = -3.4e38f;
    for (int c = threadIdx.x; c < CH_; c += 256) m = fmaxf(m, partM[(size_t)r * CH_ + c]);
    red[threadIdx.x] = m; __syncthreads();
    for (int s = 128; s > 0; s >>= 1) {
        if ((int)threadIdx.x < s) red[threadIdx.x] = fmaxf(red[threadIdx.x], red[threadIdx.x + s]);
        __syncthreads();
    }
    float M = red[0];
    __syncthreads();
    // sum exp
    float se = 0.f;
    for (int c = threadIdx.x; c < CH_; c += 256)
        se += partS[(size_t)r * CH_ + c] * expf(partM[(size_t)r * CH_ + c] - M);
    red[threadIdx.x] = se; __syncthreads();
    for (int s = 128; s > 0; s >>= 1) {
        if ((int)threadIdx.x < s) red[threadIdx.x] += red[threadIdx.x + s];
        __syncthreads();
    }
    if (threadIdx.x == 0) yp[r] = logit - (M + logf(red[0]));
}

// ---------------------------------------------------------------------------
extern "C" void kernel_launch(void* const* d_in, const int* in_sizes, int n_in,
                              void* d_out, int out_size, void* d_ws, size_t ws_size,
                              hipStream_t stream)
{
    (void)in_sizes; (void)n_in; (void)out_size; (void)ws_size;
    const int*   zi      = (const int*)  d_in[0];
    const int*   y       = (const int*)  d_in[1];
    const float* noise   = (const float*)d_in[2];
    const float* latent  = (const float*)d_in[3];
    const float* trans_w = (const float*)d_in[4];
    const float* trans_b = (const float*)d_in[5];
    const float* ek_w    = (const float*)d_in[6];
    const float* ek_b    = (const float*)d_in[7];
    const float* ev_w    = (const float*)d_in[8];
    const float* ev_b    = (const float*)d_in[9];
    const float* vocab_w = (const float*)d_in[10];
    const float* vocab_b = (const float*)d_in[11];
    float* yp = (float*)d_out;

    char* base = (char*)d_ws; size_t off = 0;
    auto alloc = [&](size_t bytes) -> void* {
        void* p = base + off;
        off = (off + bytes + 255) & ~(size_t)255;
        return p;
    };
    float* z0f     = (float*)alloc((size_t)B_ * E_ * 4);
    float* zf      = (float*)alloc((size_t)B_ * E_ * 4);
    u16*   z0b     = (u16*)  alloc((size_t)B_ * E_ * 2);
    u16*   ekwb    = (u16*)  alloc((size_t)M_ * E_ * E_ * 2);   // 32MB
    u16*   evwb    = (u16*)  alloc((size_t)M_ * E_ * E_ * 2);   // 32MB
    u16*   vwb     = (u16*)  alloc((size_t)G_ * E_ * 2);        // 64MB
    u16*   twb     = (u16*)  alloc((size_t)E_ * E_ * 2);        // 2MB
    float* ek_raw  = (float*)alloc((size_t)B_ * M_ * E_ * 4);   // 4MB
    float* ev_raw  = (float*)alloc((size_t)B_ * M_ * E_ * 4);   // 4MB
    u16*   ekb     = (u16*)  alloc((size_t)B_ * M_ * E_ * 2);   // 2MB
    float* evf     = (float*)alloc((size_t)B_ * M_ * E_ * 4);   // 4MB
    float* zsf     = (float*)alloc((size_t)B_ * NS_ * E_ * 4);  // 8MB
    u16*   zsb     = (u16*)  alloc((size_t)B_ * NS_ * E_ * 2);  // 4MB
    float* partM   = (float*)alloc((size_t)B_ * NS_ * CH_ * 4); // 1MB
    float* partS   = (float*)alloc((size_t)B_ * NS_ * CH_ * 4); // 1MB

    // 1) gather + stdnorm + noise
    prep_kernel<<<B_, 256, 0, stream>>>(zi, noise, latent, z0f, zf, z0b);

    // 2) weight conversions f32 -> bf16
    cvt_kernel<<<4096, 256, 0, stream>>>(ek_w,    ekwb, (size_t)M_ * E_ * E_);
    cvt_kernel<<<4096, 256, 0, stream>>>(ev_w,    evwb, (size_t)M_ * E_ * E_);
    cvt_kernel<<<4096, 256, 0, stream>>>(vocab_w, vwb,  (size_t)G_ * E_);
    cvt_kernel<<<1024, 256, 0, stream>>>(trans_w, twb,  (size_t)E_ * E_);

    // 3) ek / ev projections (64 x 16384, K=1024) via WMMA
    gemm_bias_kernel<<<dim3((M_ * E_) / 256, 1), 512, 0, stream>>>(
        z0b, ekwb, ek_b, ek_raw, E_, M_ * E_);
    gemm_bias_kernel<<<dim3((M_ * E_) / 256, 1), 512, 0, stream>>>(
        z0b, evwb, ev_b, ev_raw, E_, M_ * E_);

    // 4) row stdnorm: ek -> bf16 (attention), ev -> f32 (emit)
    stdnorm_kernel<<<B_ * M_, 256, 0, stream>>>(ek_raw, nullptr, ekb);
    stdnorm_kernel<<<B_ * M_, 256, 0, stream>>>(ev_raw, evf, nullptr);

    // 5) recurrence: 32 steps internal, per-batch-row independent
    recurrence_kernel<<<B_ / 16, 512, 0, stream>>>(
        zf, ekb, evf, twb, trans_b, zsf, zsb);

    // 6) vocab GEMM + fused log-softmax partials (128-row tiles)
    vocab_kernel<<<dim3(CH_, (B_ * NS_) / 128), 512, 0, stream>>>(
        zsb, vwb, vocab_b, partM, partS);

    // 7) finalize
    finalize_kernel<<<B_ * NS_, 256, 0, stream>>>(
        zsf, vocab_w, vocab_b, y, partM, partS, yp);
}
